// CMSDecoder_63050119905757
// MI455X (gfx1250) — matrix operand order repaired and checked by reading, not compiled, not run on hardware
//
#include <hip/hip_runtime.h>

typedef __attribute__((ext_vector_type(16))) _Float16 v16h;
typedef __attribute__((ext_vector_type(8)))  float    v8f;

#define NPIX   16384      // H*W
#define CCH    256        // C
#define NBATCH 16         // B
#define TILE   128        // pixels per block in GEMM kernels

// ---------------------------------------------------------------------------
// Async global->LDS copies (CDNA5 GLOBAL_LOAD_ASYNC_TO_LDS_B128, ASYNCcnt).
// Probe (round 2): builtin expects (v4i __device__*, v4i __shared__*, imm, imm)
// with v4i = int __attribute__((vector_size(16))).
// ---------------------------------------------------------------------------
#if defined(__has_builtin)
#if __has_builtin(__builtin_amdgcn_global_load_async_to_lds_b128) && \
    __has_builtin(__builtin_amdgcn_s_wait_asynccnt)
#define HAVE_ASYNC 1
#endif
#endif

#if HAVE_ASYNC
typedef int v4i __attribute__((vector_size(16)));
typedef __attribute__((address_space(1))) v4i* as1p;
typedef __attribute__((address_space(3))) v4i* as3p;
__device__ __forceinline__ void async_b128(const void* g, void* l) {
  __builtin_amdgcn_global_load_async_to_lds_b128((as1p)g, (as3p)l, 0, 0);
}
#define WAITA(n) __builtin_amdgcn_s_wait_asynccnt(n)
#else
#define WAITA(n)
#endif

// Copy one [nrows][32]-f16 K-chunk of a row-major [nrows][ldh] f16 matrix
// into LDS. Async path: one b128 lane-transfer per 8 halves.
__device__ __forceinline__ void load_w_chunk(const _Float16* __restrict__ g,
                                             int ldh, _Float16* l, int nrows,
                                             int k0, int t) {
#if HAVE_ASYNC
  for (int i = t; i < nrows * 4; i += 512) {
    int rr = i >> 2, seg = (i & 3) * 8;
    async_b128(g + rr * ldh + k0 + seg, l + rr * 32 + seg);
  }
#else
  for (int i = t; i < nrows * 32; i += 512)
    l[i] = g[(i >> 5) * ldh + k0 + (i & 31)];
#endif
}

// ---------------------------------------------------------------------------
// WMMA 16x16x32 f16 fragment gathers from LDS (row-major, K contiguous).
// A (ISA 7.12.2): VGPR j holds K = (j<4?0:16) + hi*8 + (j&3)*2.
// B (same as A): VGPR j holds K = (j<4?0:16) + hi*8 + (j&3)*2.
// Pairs contiguous -> compiler merges into ds_load_b128 (verified round 1).
// ---------------------------------------------------------------------------
__device__ __forceinline__ v16h frag_a(const _Float16* sh, int row0, int ldk,
                                       int k0, int lane) {
  int m  = row0 + (lane & 15);
  int hi = lane >> 4;
  const _Float16* p = sh + m * ldk + k0;
  union { v16h v; unsigned u[8]; } r;
#pragma unroll
  for (int j = 0; j < 8; ++j) {
    int kb = ((j & 4) << 2) + hi * 8 + ((j & 3) << 1);
    r.u[j] = *(const unsigned*)(p + kb);
  }
  return r.v;
}

__device__ __forceinline__ v16h frag_b(const _Float16* sh, int col0, int ldk,
                                       int k0, int lane) {
  int n  = col0 + (lane & 15);
  int hi = lane >> 4;
  const _Float16* p = sh + n * ldk + k0;
  union { v16h v; unsigned u[8]; } r;
#pragma unroll
  for (int j = 0; j < 8; ++j) {
    int kb = ((j & 4) << 2) + hi * 8 + ((j & 3) << 1);
    r.u[j] = *(const unsigned*)(p + kb);
  }
  return r.v;
}

__device__ __forceinline__ float phi(float x) {          // elu(x)+1
  return x > 0.f ? x + 1.f : __expf(x);
}

// ---------------------------------------------------------------------------
// Kernel 0: one-time weight conversion fp32 -> f16 into workspace.
//   wkv: [128][256] (Wk rows 0-63, Wv rows 64-127), wq: [64][256], wo: [256][64]
// ---------------------------------------------------------------------------
__global__ __launch_bounds__(256) void k_wcvt(
    const float* __restrict__ Wk, const float* __restrict__ Wv,
    const float* __restrict__ Wq, const float* __restrict__ Wo,
    _Float16* __restrict__ wkv, _Float16* __restrict__ wq,
    _Float16* __restrict__ wo) {
  int i = blockIdx.x * 256 + threadIdx.x;          // 32768 threads
  if (i < 16384) {
    wkv[i]         = (_Float16)Wk[i];
    wkv[i + 16384] = (_Float16)Wv[i];
    wq[i]          = (_Float16)Wq[i];
    wo[i]          = (_Float16)Wo[i];
  }
}

// ---------------------------------------------------------------------------
// Kernel 1: per-(b,c) channel sum / sum-of-squares (one block per channel)
// ---------------------------------------------------------------------------
__global__ __launch_bounds__(256) void k_chansum(const float* __restrict__ x,
                                                 float* __restrict__ csum,
                                                 float* __restrict__ csum2) {
  int bc = blockIdx.x;
  const float4* p = (const float4*)(x + (size_t)bc * NPIX);
  float s = 0.f, s2 = 0.f;
  for (int i = threadIdx.x; i < NPIX / 4; i += 256) {
    float4 v = p[i];
    s  += v.x + v.y + v.z + v.w;
    s2 += v.x * v.x + v.y * v.y + v.z * v.z + v.w * v.w;
  }
  __shared__ float rs[256], rs2[256];
  int t = threadIdx.x;
  rs[t] = s; rs2[t] = s2;
  __syncthreads();
  for (int off = 128; off > 0; off >>= 1) {
    if (t < off) { rs[t] += rs[t + off]; rs2[t] += rs2[t + off]; }
    __syncthreads();
  }
  if (t == 0) { csum[bc] = rs[0]; csum2[bc] = rs2[0]; }
}

// ---------------------------------------------------------------------------
// Kernel 2: per-batch group-norm affine (scale/shift), gate MLP, M/Z init
// ---------------------------------------------------------------------------
__global__ __launch_bounds__(256) void k_prep(
    const float* __restrict__ csum, const float* __restrict__ csum2,
    const float* __restrict__ gn_w, const float* __restrict__ gn_b,
    const float* __restrict__ M0,   const float* __restrict__ Z0,
    const float* __restrict__ g1w,  const float* __restrict__ g1b,
    const float* __restrict__ g2w,  const float* __restrict__ g2b,
    float* __restrict__ scale, float* __restrict__ shift,
    float* __restrict__ Mws,   float* __restrict__ Zws,
    float* __restrict__ gatews, float* __restrict__ gate_out) {
  int b = blockIdx.x, t = threadIdx.x;
  __shared__ float gsum[8], gsum2[8], pooled[256], gact[64];

  if (t < 8) {
    float s = 0.f, s2 = 0.f;
    for (int j = 0; j < 32; ++j) {
      s  += csum [b * CCH + t * 32 + j];
      s2 += csum2[b * CCH + t * 32 + j];
    }
    gsum[t] = s; gsum2[t] = s2;
  }
  __syncthreads();

  const float cnt = 32.f * (float)NPIX;
  int c = t, g = c >> 5;
  float mu   = gsum[g] / cnt;
  float var  = gsum2[g] / cnt - mu * mu;
  float rsig = rsqrtf(var + 1e-5f);
  float sc = rsig * gn_w[c];
  float sh = gn_b[c] - mu * sc;
  scale[b * CCH + c] = sc;
  shift[b * CCH + c] = sh;
  pooled[c] = (csum[b * CCH + c] / (float)NPIX) * sc + sh;
  __syncthreads();

  if (t < 64) {
    float a = g1b[t];
    for (int j = 0; j < CCH; ++j) a += g1w[t * CCH + j] * pooled[j];
    gact[t] = 0.5f * a * (1.f + erff(a * 0.70710678118654752f));  // exact GELU
  }
  __syncthreads();
  if (t == 0) {
    float z = g2b[0];
    for (int j = 0; j < 64; ++j) z += g2w[j] * gact[j];
    float gt = 1.f / (1.f + __expf(-z));
    gatews[b] = gt;
    gate_out[b] = gt;
  }
  for (int i = t; i < 1024; i += 256) Mws[b * 1024 + i] = M0[i];  // warm start
  if (t < 64) Zws[b * 64 + t] = Z0[t];
}

// ---------------------------------------------------------------------------
// Kernel 3: per 128-pixel tile, k/v GEMM (WMMA, async-staged weights),
//           M += phi(k) v^T (WMMA), Z += sum phi(k); global f32 atomics.
// grid (128, B), 512 threads = 16 waves
// ---------------------------------------------------------------------------
__global__ __launch_bounds__(512) void k_accum_MZ(
    const float* __restrict__ x, const _Float16* __restrict__ wkv,
    const float* __restrict__ scale, const float* __restrict__ shift,
    float* __restrict__ Mws, float* __restrict__ Zws) {
  __shared__ alignas(16) char smem[16384 * 2 + 4096 + 256];
  _Float16* wh0 = (_Float16*)smem;              // [128][32] weight buf 0
  _Float16* wh1 = (_Float16*)(smem + 8192);     // [128][32] weight buf 1
  _Float16* xh  = (_Float16*)(smem + 16384);    // [128 px][32 ch]
  _Float16* pk  = (_Float16*)smem;              // [64][128] phi(k) (post-loop)
  _Float16* vv  = (_Float16*)(smem + 16384);    // [64][128] v      (post-loop)
  float*    Mt  = (float*)(smem + 32768);       // [4][16][16]
  float*    Zl  = (float*)(smem + 32768 + 4096);// [64]

  int b = blockIdx.y;
  int n0 = blockIdx.x * TILE;
  int t = threadIdx.x, w = t >> 5, lane = t & 31;
  int cg = w & 7, rg = (w >> 3) * 4;            // 8 col tiles x 8 row tiles
  const float* sc = scale + b * CCH;
  const float* sf = shift + b * CCH;

  v8f acc[4] = {};
  load_w_chunk(wkv, CCH, wh0, 128, 0, t);       // prefetch chunk 0
  for (int kc = 0; kc < 8; ++kc) {              // K = 256 in chunks of 32
    int k0 = kc * 32;
    __syncthreads();                            // (a) prev reads of xh done
    {
      int px = t & 127, cb = (t >> 7) * 8;
#pragma unroll
      for (int j = 0; j < 8; ++j) {
        int ch = k0 + cb + j;
        float xv = x[((size_t)b * CCH + ch) * NPIX + n0 + px];
        xh[px * 32 + cb + j] = (_Float16)(xv * sc[ch] + sf[ch]);
      }
    }
    if (kc < 7) {
      load_w_chunk(wkv, CCH, (kc & 1) ? wh0 : wh1, 128, k0 + 32, t);
      WAITA(1);
    } else {
      WAITA(0);
    }
    __syncthreads();                            // (b) xh + weights visible
    v16h bf = frag_b(xh, cg * 16, 32, 0, lane);
    const _Float16* wcur = (kc & 1) ? wh1 : wh0;
#pragma unroll
    for (int ti = 0; ti < 4; ++ti) {
      v16h af = frag_a(wcur, (rg + ti) * 16, 32, 0, lane);
      acc[ti] = __builtin_amdgcn_wmma_f32_16x16x32_f16(
          false, af, false, bf, (short)0, acc[ti], false, false);
    }
  }
  __syncthreads();

  // scatter phi(k) / v tiles to LDS, zero reduction buffers
#pragma unroll
  for (int ti = 0; ti < 4; ++ti) {
    int px = cg * 16 + (lane & 15);
#pragma unroll
    for (int r = 0; r < 8; ++r) {
      int row = (rg + ti) * 16 + r + (lane >> 4) * 8;
      float val = acc[ti][r];
      if (row < 64) pk[row * TILE + px] = (_Float16)phi(val);
      else          vv[(row - 64) * TILE + px] = (_Float16)val;
    }
  }
  for (int i = t; i < 1024; i += 512) Mt[i] = 0.f;
  if (t < 64) Zl[t] = 0.f;
  __syncthreads();

  {  // Z partials
    int row = t & 63, s0 = (t >> 6) * 16;
    float s = 0.f;
#pragma unroll
    for (int j = 0; j < 16; ++j) s += (float)pk[row * TILE + s0 + j];
    atomicAdd(&Zl[row], s);
  }
  {  // M partials: head = w&3, pixel chunk = w>>2
    int h = w & 3, pc = w >> 2;
    v16h af = frag_a(pk, h * 16, TILE, pc * 32, lane);
    v16h bf = frag_b(vv, h * 16, TILE, pc * 32, lane);
    v8f macc = {};
    macc = __builtin_amdgcn_wmma_f32_16x16x32_f16(
        false, af, false, bf, (short)0, macc, false, false);
#pragma unroll
    for (int r = 0; r < 8; ++r) {
      int d = r + (lane >> 4) * 8, e = lane & 15;
      atomicAdd(&Mt[h * 256 + d * 16 + e], macc[r]);
    }
  }
  __syncthreads();
  for (int i = t; i < 1024; i += 512) atomicAdd(&Mws[b * 1024 + i], Mt[i]);
  if (t < 64) atomicAdd(&Zws[b * 64 + t], Zl[t]);
}

// ---------------------------------------------------------------------------
// Kernel 4: q GEMM (WMMA) -> phi(q) -> y = (M phi_q)/max(Z phi_q, 1e-4)
//           -> Wo GEMM (WMMA, async-staged) -> out = x + gate*(y+bo)
// grid (128, B), 512 threads
// ---------------------------------------------------------------------------
__global__ __launch_bounds__(512) void k_output(
    const float* __restrict__ x, const _Float16* __restrict__ wq,
    const _Float16* __restrict__ wo, const float* __restrict__ bo,
    const float* __restrict__ scale, const float* __restrict__ shift,
    const float* __restrict__ Mws, const float* __restrict__ Zws,
    const float* __restrict__ gatews, float* __restrict__ out) {
  __shared__ alignas(16) char smem[53504];
  _Float16* wqh0 = (_Float16*)smem;             // [64][32]  phase 1 buf 0
  _Float16* wqh1 = (_Float16*)(smem + 4096);    // [64][32]  phase 1 buf 1
  _Float16* xh   = (_Float16*)(smem + 8192);    // [128][32] phase 1
  float*    phiQ = (float*)smem;                // [64][128] phase 2 (32KB)
  _Float16* yh   = (_Float16*)(smem + 32768);   // [128 px][64 ch] (16KB)
  float*    Ml   = (float*)(smem + 49152);      // [4][16][16]
  float*    Zlds = (float*)(smem + 53248);      // [64]
  _Float16* woh0 = (_Float16*)smem;             // [256][32] phase 3 buf 0
  _Float16* woh1 = (_Float16*)(smem + 16384);   // [256][32] phase 3 buf 1

  int b = blockIdx.y;
  int n0 = blockIdx.x * TILE;
  int t = threadIdx.x, w = t >> 5, lane = t & 31;
  int cg = w & 7;
  const float* sc = scale + b * CCH;
  const float* sf = shift + b * CCH;
  float gate = gatews[b];

  // ---- phase 1: q = Wq @ x_norm (64 x 128 tile) -------------------------
  int rgq = (w >> 3) * 2;
  v8f accq[2] = {};
  load_w_chunk(wq, CCH, wqh0, 64, 0, t);
  for (int kc = 0; kc < 8; ++kc) {
    int k0 = kc * 32;
    __syncthreads();
    {
      int px = t & 127, cb = (t >> 7) * 8;
#pragma unroll
      for (int j = 0; j < 8; ++j) {
        int ch = k0 + cb + j;
        float xv = x[((size_t)b * CCH + ch) * NPIX + n0 + px];
        xh[px * 32 + cb + j] = (_Float16)(xv * sc[ch] + sf[ch]);
      }
    }
    if (kc < 7) {
      load_w_chunk(wq, CCH, (kc & 1) ? wqh0 : wqh1, 64, k0 + 32, t);
      WAITA(1);
    } else {
      WAITA(0);
    }
    __syncthreads();
    v16h bf = frag_b(xh, cg * 16, 32, 0, lane);
    const _Float16* wcur = (kc & 1) ? wqh1 : wqh0;
#pragma unroll
    for (int ti = 0; ti < 2; ++ti) {
      v16h af = frag_a(wcur, (rgq + ti) * 16, 32, 0, lane);
      accq[ti] = __builtin_amdgcn_wmma_f32_16x16x32_f16(
          false, af, false, bf, (short)0, accq[ti], false, false);
    }
  }
  __syncthreads();

  // ---- phase 2: phi(q) to LDS; load M/Z; y to LDS -----------------------
#pragma unroll
  for (int ti = 0; ti < 2; ++ti) {
    int px = cg * 16 + (lane & 15);
#pragma unroll
    for (int r = 0; r < 8; ++r) {
      int row = (rgq + ti) * 16 + r + (lane >> 4) * 8;
      phiQ[row * TILE + px] = phi(accq[ti][r]);
    }
  }
  for (int i = t; i < 1024; i += 512) Ml[i] = Mws[b * 1024 + i];
  if (t < 64) Zlds[t] = Zws[b * 64 + t];
  __syncthreads();

  for (int i = t; i < 64 * TILE; i += 512) {
    int row = i & 63, px = i >> 6;
    int h = row >> 4, e = row & 15;
    const float* Mh = Ml + h * 256;
    const float* Zh = Zlds + h * 16;
    const float* pq = phiQ + (h * 16) * TILE + px;
    float num = 0.f, den = 0.f;
#pragma unroll
    for (int d = 0; d < 16; ++d) {
      float p = pq[d * TILE];
      num += Mh[d * 16 + e] * p;
      den += Zh[d] * p;
    }
    den = fmaxf(den, 1e-4f);
    yh[px * 64 + row] = (_Float16)(num / den);
  }
  __syncthreads();                              // y done; phiQ region now free

  // ---- phase 3: out = Wo @ y (256 x 128), fused residual + gate ---------
  int rgo = (w >> 3) * 8;
  v8f acco[8] = {};
  load_w_chunk(wo, 64, woh0, 256, 0, t);
  load_w_chunk(wo, 64, woh1, 256, 32, t);
  WAITA(2);
  __syncthreads();
  {
    v16h bf = frag_b(yh, cg * 16, 64, 0, lane);
#pragma unroll
    for (int ti = 0; ti < 8; ++ti) {
      v16h af = frag_a(woh0, (rgo + ti) * 16, 32, 0, lane);
      acco[ti] = __builtin_amdgcn_wmma_f32_16x16x32_f16(
          false, af, false, bf, (short)0, acco[ti], false, false);
    }
  }
  WAITA(0);
  __syncthreads();
  {
    v16h bf = frag_b(yh, cg * 16, 64, 32, lane);
#pragma unroll
    for (int ti = 0; ti < 8; ++ti) {
      v16h af = frag_a(woh1, (rgo + ti) * 16, 32, 0, lane);
      acco[ti] = __builtin_amdgcn_wmma_f32_16x16x32_f16(
          false, af, false, bf, (short)0, acco[ti], false, false);
    }
  }

  int pxg = n0 + cg * 16 + (lane & 15);
#pragma unroll
  for (int ti = 0; ti < 8; ++ti) {
#pragma unroll
    for (int r = 0; r < 8; ++r) {
      int row = (rgo + ti) * 16 + r + (lane >> 4) * 8;
      size_t idx = ((size_t)b * CCH + row) * NPIX + pxg;
      out[idx] = x[idx] + gate * (acco[ti][r] + bo[row]);
    }
  }
}

// ---------------------------------------------------------------------------
// host launcher
// ---------------------------------------------------------------------------
extern "C" void kernel_launch(void* const* d_in, const int* in_sizes, int n_in,
                              void* d_out, int out_size, void* d_ws,
                              size_t ws_size, hipStream_t stream) {
  const float* x    = (const float*)d_in[0];
  const float* gn_w = (const float*)d_in[1];
  const float* gn_b = (const float*)d_in[2];
  const float* Wk   = (const float*)d_in[3];
  const float* Wv   = (const float*)d_in[4];
  const float* Wq   = (const float*)d_in[5];
  const float* Wo   = (const float*)d_in[6];
  const float* bo   = (const float*)d_in[7];
  const float* M0   = (const float*)d_in[8];
  const float* Z0   = (const float*)d_in[9];
  const float* g1w  = (const float*)d_in[10];
  const float* g1b  = (const float*)d_in[11];
  const float* g2w  = (const float*)d_in[12];
  const float* g2b  = (const float*)d_in[13];

  float* ws = (float*)d_ws;
  float* csum   = ws;                      // B*C
  float* csum2  = csum  + NBATCH * CCH;
  float* scale  = csum2 + NBATCH * CCH;
  float* shift  = scale + NBATCH * CCH;
  float* gatews = shift + NBATCH * CCH;
  float* Mws    = gatews + NBATCH;         // B*4*16*16
  float* Zws    = Mws + NBATCH * 1024;     // B*64
  _Float16* wkvh = (_Float16*)(Zws + NBATCH * 64);  // [128][256]
  _Float16* wqh  = wkvh + 128 * CCH;                // [64][256]
  _Float16* woh  = wqh  + 64 * CCH;                 // [256][64]

  float* out      = (float*)d_out;
  float* gate_out = out + (size_t)NBATCH * CCH * NPIX;

  k_wcvt<<<128, 256, 0, stream>>>(Wk, Wv, Wq, Wo, wkvh, wqh, woh);
  k_chansum<<<NBATCH * CCH, 256, 0, stream>>>(x, csum, csum2);
  k_prep<<<NBATCH, 256, 0, stream>>>(csum, csum2, gn_w, gn_b, M0, Z0, g1w, g1b,
                                     g2w, g2b, scale, shift, Mws, Zws, gatews,
                                     gate_out);
  dim3 grid(NPIX / TILE, NBATCH);
  k_accum_MZ<<<grid, 512, 0, stream>>>(x, wkvh, scale, shift, Mws, Zws);
  k_output<<<grid, 512, 0, stream>>>(x, wqh, woh, bo, scale, shift, Mws, Zws,
                                     gatews, out);
}
